// NRU_1211180777757
// MI455X (gfx1250) — compile-verified, matrix operand order checked by reading, not run on hardware
//
#include <hip/hip_runtime.h>
#include <cmath>

typedef _Float16 v16h __attribute__((ext_vector_type(16)));
typedef _Float16 v8h  __attribute__((ext_vector_type(8)));
typedef float    v8f  __attribute__((ext_vector_type(8)));
typedef int      i32x4 __attribute__((vector_size(16)));

#define S_LEN   512
#define BATCH   128
#define IN_D    256
#define HID_D   512
#define MEM_D   768
#define CK      1536      // A width in K order [h(512) | m(768) | x(256)]
#define HMK     1280      // h+m prefix (the "hm" GEMM K)
#define A_STRIDE 1544     // halves per A row (+8 pad -> conflict-free LDS)
#define NT_MAIN 32        // 512/16 N tiles, main GEMM
#define KT_MAIN 48        // 1536/32 K steps
#define NS      224       // packed small-GEMM N (96 va | 96 vb | 16 al | 16 be)
#define NT_S    14
#define KT_S    40        // 1280/32
#define BT      16        // batch rows per workgroup

// Fixed LDS layout (host and device must agree -> constants, not #if'd)
#define LDS_A_OFF    0
#define LDS_UAB_OFF  (BT * A_STRIDE * 2)                    // 49408
#define LDS_MST_OFF  (LDS_UAB_OFF + BT * NS * 4)            // 63744
#define LDS_XBUF_OFF (LDS_MST_OFF + BT * MEM_D * 4)         // 112896
#define LDS_TOTAL    (LDS_XBUF_OFF + 2 * BT * IN_D * 4)     // 145664

// CDNA5 async global->LDS staging (guarded; falls back to plain loads)
#if defined(__HIP_DEVICE_COMPILE__) && \
    __has_builtin(__builtin_amdgcn_global_load_async_to_lds_b128) && \
    __has_builtin(__builtin_amdgcn_s_wait_asynccnt)
#define NRU_ASYNC 1
#else
#define NRU_ASYNC 0
#endif

// ---------------- helpers ----------------

// (lane,half)->k mapping shared by WMMA A and B 16-bit fragments:
//   k_local = (e&7) + (e>>3)*16 + (lane>>4)*8
__device__ __forceinline__ v16h load_a_frag(const _Float16* A, int lane, int kt) {
  const int m   = lane & 15;
  const int grp = lane >> 4;
  const _Float16* p = A + m * A_STRIDE + kt * 32 + grp * 8;
  v8h lo = *(const v8h*)(p);        // k_local [grp*8 .. grp*8+7]
  v8h hi = *(const v8h*)(p + 16);   // k_local [16+grp*8 .. 16+grp*8+7]
  return __builtin_shufflevector(lo, hi, 0,1,2,3,4,5,6,7,8,9,10,11,12,13,14,15);
}

__device__ __forceinline__ void fifth_split(float x, float& p, float& n) {
  float a  = fabsf(x);
  float x2 = x * x;
  float f5 = x2 * x2 * a;          // |x|^5
  p = (x > 0.f) ? f5 : 0.f;
  n = (x < 0.f) ? f5 : 0.f;
}

#if NRU_ASYNC
// DMA one step's x tile (16 rows x 256 f32 = 16KB = 1024 x b128) into LDS.
// 4 async-b128 per thread; one ASYNCcnt increment per wave per issue.
__device__ __forceinline__ void async_x_load(const float* xp, float* xb, int tid) {
  #pragma unroll
  for (int i = 0; i < 4; ++i) {
    int c = i * 256 + tid;   // 16-byte chunk id
    i32x4 __attribute__((address_space(1)))* g =
        (i32x4 __attribute__((address_space(1)))*)(uintptr_t)(xp + c * 4);
    i32x4 __attribute__((address_space(3)))* l =
        (i32x4 __attribute__((address_space(3)))*)(uint32_t)(uintptr_t)(xb + c * 4);
    __builtin_amdgcn_global_load_async_to_lds_b128(g, l, 0, 0);
  }
}
#endif

// ---------------- one-time weight packing ----------------

// W_h (512 x 1536, input order [x|h|m]) -> f16 B fragments, K permuted to [h|m|x].
__global__ __launch_bounds__(256) void pack_wh(const float* __restrict__ Wh,
                                               _Float16* __restrict__ o) {
  int tid = blockIdx.x * 256 + threadIdx.x;     // 48*32*512 = 786432 total
  int e  = tid & 15;
  int l  = (tid >> 4) & 31;
  int f  = tid >> 9;
  int kt = f >> 5;
  int nt = f & 31;
  int n  = nt * 16 + (l & 15);
  int kl = (e & 7) + ((e >> 3) << 4) + ((l >> 4) << 3);
  int k  = kt * 32 + kl;
  int orig = (k < HMK) ? (k + IN_D) : (k - HMK); // [h|m|x] -> original [x|h|m]
  o[tid] = (_Float16)Wh[n * CK + orig];
}

// W_va/W_vb/W_al/W_be -> single f16 B matrix, N=224, K=1280 (order already [h|m]).
__global__ __launch_bounds__(256) void pack_ws(const float* __restrict__ Wva,
                                               const float* __restrict__ Wvb,
                                               const float* __restrict__ Wal,
                                               const float* __restrict__ Wbe,
                                               _Float16* __restrict__ o) {
  int tid = blockIdx.x * 256 + threadIdx.x;     // 40*14*512 = 286720 total
  if (tid >= KT_S * NT_S * 512) return;
  int e  = tid & 15;
  int l  = (tid >> 4) & 31;
  int f  = tid >> 9;
  int kt = f / NT_S;
  int nt = f % NT_S;
  int n  = nt * 16 + (l & 15);
  int kl = (e & 7) + ((e >> 3) << 4) + ((l >> 4) << 3);
  int k  = kt * 32 + kl;
  float v;
  if      (n < 96)  v = Wva[n * HMK + k];
  else if (n < 192) v = Wvb[(n - 96) * HMK + k];
  else if (n < 208) { int r = n - 192; v = (r < 3) ? Wal[r * HMK + k] : 0.f; }
  else              { int r = n - 208; v = (r < 3) ? Wbe[r * HMK + k] : 0.f; }
  o[tid] = (_Float16)v;
}

__global__ void pack_bias(const float* __restrict__ bva, const float* __restrict__ bvb,
                          const float* __restrict__ bal, const float* __restrict__ bbe,
                          float* __restrict__ o) {
  int n = threadIdx.x;
  if (n >= NS) return;
  float v;
  if      (n < 96)  v = bva[n];
  else if (n < 192) v = bvb[n - 96];
  else if (n < 208) { int r = n - 192; v = (r < 3) ? bal[r] : 0.f; }
  else              { int r = n - 208; v = (r < 3) ? bbe[r] : 0.f; }
  o[n] = v;
}

// ---------------- main recurrent kernel ----------------
// grid = 8 workgroups (16 batch rows each), block = 256 threads = 8 wave32.
__global__ __launch_bounds__(256)
void nru_main(const float* __restrict__ x,
              const _Float16* __restrict__ Wh,   // packed fragments
              const float* __restrict__ bh,
              const _Float16* __restrict__ Ws,   // packed fragments
              const float* __restrict__ bs,      // packed bias (224)
              float* __restrict__ out) {
  extern __shared__ char smem[];
  _Float16* A   = (_Float16*)(smem + LDS_A_OFF);   // 16 x 1544 f16
  float*    uab = (float*)(smem + LDS_UAB_OFF);    // 16 x 224 f32
  float*    mst = (float*)(smem + LDS_MST_OFF);    // 16 x 768 f32
#if NRU_ASYNC
  float*    xbf = (float*)(smem + LDS_XBUF_OFF);   // 2 x (16 x 256) f32
#endif

  const int tid  = threadIdx.x;
  const int lane = tid & 31;
  const int wv   = tid >> 5;
  const int bb   = blockIdx.x * BT;

  float* hn_out = out + (size_t)S_LEN * BATCH * HID_D;
  float* mn_out = hn_out + (size_t)BATCH * HID_D;

  // h = 0, m = 0 (A cols [0,1280)), m-state = 0
  for (int i = tid; i < BT * HMK; i += 256)
    A[(i / HMK) * A_STRIDE + (i % HMK)] = (_Float16)0.f;
  for (int i = tid; i < BT * MEM_D; i += 256) mst[i] = 0.f;

#if NRU_ASYNC
  // preload x_0 into buffer 0 (ASYNCcnt = 4 per wave)
  async_x_load(x + (size_t)bb * IN_D, xbf, tid);
#endif
  __syncthreads();

  for (int t = 0; t < S_LEN; ++t) {
    // ---- Phase 1: stage x_t -> f16 into A cols [1280,1536) ----
#if NRU_ASYNC
    {
      if (t + 1 < S_LEN) {
        // kick off next step's DMA, then drain down to it (oldest 4 = step t done;
        // ASYNCcnt completes in issue order per ISA)
        async_x_load(x + ((size_t)(t + 1) * BATCH + bb) * IN_D,
                     xbf + ((t + 1) & 1) * (BT * IN_D), tid);
        __builtin_amdgcn_s_wait_asynccnt(4);
      } else {
        __builtin_amdgcn_s_wait_asynccnt(0);
      }
      // convert exactly the floats this thread DMA'd (own-wave wait suffices)
      const float* xs = xbf + (t & 1) * (BT * IN_D);
      #pragma unroll
      for (int i = 0; i < 4; ++i) {
        int c = i * 256 + tid;          // 16B chunk -> 4 consecutive floats
        int r = c >> 6;                 // (c*4) >> 8
        int col = (c * 4) & 255;
        #pragma unroll
        for (int u = 0; u < 4; ++u)
          A[r * A_STRIDE + HMK + col + u] = (_Float16)xs[c * 4 + u];
      }
    }
#else
    {
      const float* xp = x + ((size_t)t * BATCH + bb) * IN_D;
      #pragma unroll
      for (int i = 0; i < 16; ++i) {
        int e = i * 256 + tid;
        int r = e >> 8, c = e & 255;
        A[r * A_STRIDE + HMK + c] = (_Float16)xp[r * IN_D + c];
        if (t + 1 < S_LEN)
          __builtin_prefetch(xp + (size_t)BATCH * IN_D + r * IN_D + c, 0, 0);
      }
    }
#endif
    __syncthreads();

    // ---- Phase 2: main GEMM h_new = relu([h|m|x] * Wh^T + bh) ----
    v8f acc[4];
    {
      v8f z = {0.f,0.f,0.f,0.f,0.f,0.f,0.f,0.f};
      acc[0] = z; acc[1] = z; acc[2] = z; acc[3] = z;
      const int ntb = wv * 4;
      #pragma unroll 2
      for (int kt = 0; kt < KT_MAIN; ++kt) {
        v16h a = load_a_frag(A, lane, kt);
        const _Float16* wb = Wh + (((size_t)kt * NT_MAIN + ntb) * 32 + lane) * 16;
        #pragma unroll
        for (int j = 0; j < 4; ++j) {
          v16h b = *(const v16h*)(wb + (size_t)j * 512);
          acc[j] = __builtin_amdgcn_wmma_f32_16x16x32_f16(
              false, a, false, b, (short)0, acc[j], false, false);
        }
      }
    }
    __syncthreads();   // all waves finished reading h_old/m/x from A

    // ---- Phase 3: bias+relu; h_new -> A (f16) and out (f32) ----
    {
      float* outp = out + ((size_t)t * BATCH + bb) * HID_D;
      const int grp = lane >> 4, nl = lane & 15;
      #pragma unroll
      for (int j = 0; j < 4; ++j) {
        int n = wv * 64 + j * 16 + nl;
        float bias = bh[n];
        #pragma unroll
        for (int r = 0; r < 8; ++r) {
          int m = r + grp * 8;
          float v = fmaxf(acc[j][r] + bias, 0.f);
          A[m * A_STRIDE + n] = (_Float16)v;
          outp[(size_t)m * HID_D + n] = v;
        }
      }
    }
    __syncthreads();

    // ---- Phase 4: small GEMM  [h_new|m] * [Wva|Wvb|Wal|Wbe]^T + bias ----
    for (int tt = wv; tt < NT_S; tt += 8) {
      v8f s = {0.f,0.f,0.f,0.f,0.f,0.f,0.f,0.f};
      #pragma unroll 2
      for (int kt = 0; kt < KT_S; ++kt) {
        v16h a = load_a_frag(A, lane, kt);
        v16h b = *(const v16h*)(Ws + (((size_t)kt * NT_S + tt) * 32 + lane) * 16);
        s = __builtin_amdgcn_wmma_f32_16x16x32_f16(
            false, a, false, b, (short)0, s, false, false);
      }
      const int grp = lane >> 4, nl = lane & 15;
      int n = tt * 16 + nl;
      float bias = bs[n];
      #pragma unroll
      for (int r = 0; r < 8; ++r)
        uab[(r + grp * 8) * NS + n] = s[r] + bias;
    }
    __syncthreads();

    // ---- Phase 5: rank-1 epilogue + L5 norm + memory update (2 rows / wave) ----
    for (int bi = 0; bi < 2; ++bi) {
      const int b = wv * 2 + bi;
      const float* row = uab + b * NS;   // [ua(96) | ub(96) | al.. | be..]
      float ca[3], cb[3];
      #pragma unroll
      for (int mat = 0; mat < 2; ++mat) {
        const int base0 = mat ? 96 : 0;
        const int base1 = base0 + 48;
        const int abase = mat ? 208 : 192;
        // sign-split |u0|^5 sums per k (3 groups of 16)
        float p_a, n_a, p_b, n_b;
        fifth_split(row[base0 + lane], p_a, n_a);
        fifth_split((lane < 16) ? row[base0 + 32 + lane] : 0.f, p_b, n_b);
        #pragma unroll
        for (int off = 8; off >= 1; off >>= 1) {
          p_a += __shfl_xor(p_a, off, 32);  n_a += __shfl_xor(n_a, off, 32);
          p_b += __shfl_xor(p_b, off, 32);  n_b += __shfl_xor(n_b, off, 32);
        }
        float p0[3], n0[3];
        p0[0] = __shfl(p_a, 0, 32);  n0[0] = __shfl(n_a, 0, 32);
        p0[1] = __shfl(p_a, 16, 32); n0[1] = __shfl(n_a, 16, 32);
        p0[2] = __shfl(p_b, 0, 32);  n0[2] = __shfl(n_b, 0, 32);
        // sign-split |u1|^5 sum (48 values, shared across k)
        float p1, n1, p1b, n1b;
        fifth_split(row[base1 + lane], p1, n1);
        fifth_split((lane < 16) ? row[base1 + 32 + lane] : 0.f, p1b, n1b);
        p1 += p1b; n1 += n1b;
        #pragma unroll
        for (int off = 16; off >= 1; off >>= 1) {
          p1 += __shfl_xor(p1, off, 32);  n1 += __shfl_xor(n1, off, 32);
        }
        float* cc = mat ? cb : ca;
        #pragma unroll
        for (int k = 0; k < 3; ++k) {
          float s5  = p0[k] * p1 + n0[k] * n1;        // sum of relu(v)^5
          float den = fmaxf(powf(s5, 0.2f), 1e-12f);  // max(||v||_5, eps)
          float act = fmaxf(row[abase + k], 0.f);     // relu(alpha/beta)
          cc[k] = act / (3.f * den);                  // includes mean over K
        }
      }
      // m_new = m + sum_k( ca_k*relu(ua0_i*ua1_j) - cb_k*relu(ub0_i*ub1_j) )
      float*     mrow = mst + b * MEM_D;
      _Float16*  arow = A + b * A_STRIDE + HID_D;     // m cols of A
      #pragma unroll 4
      for (int it = 0; it < MEM_D / 32; ++it) {
        int mem = it * 32 + lane;
        int q = mem / 48, j = mem % 48;
        float a1 = row[48 + j], b1 = row[144 + j];
        float m_acc = mrow[mem];
        #pragma unroll
        for (int k = 0; k < 3; ++k) {
          m_acc += ca[k] * fmaxf(row[k * 16 + q] * a1, 0.f);
          m_acc -= cb[k] * fmaxf(row[96 + k * 16 + q] * b1, 0.f);
        }
        mrow[mem] = m_acc;
        arow[mem] = (_Float16)m_acc;
      }
    }
    __syncthreads();
  }

  // h_n: last-step h is already in `out` (L2-resident) -> plain copy, no
  // per-element branches inside the hot loop.
  {
    const float* lasth = out + ((size_t)(S_LEN - 1) * BATCH + bb) * HID_D;
    for (int i = tid; i < BT * HID_D; i += 256)
      hn_out[(size_t)bb * HID_D + i] = lasth[i];
  }
  // m_n from LDS state
  for (int i = tid; i < BT * MEM_D; i += 256) {
    int r = i / MEM_D, c = i % MEM_D;
    mn_out[(size_t)(bb + r) * MEM_D + c] = mst[i];
  }
}

// ---------------- launch ----------------

extern "C" void kernel_launch(void* const* d_in, const int* in_sizes, int n_in,
                              void* d_out, int out_size, void* d_ws, size_t ws_size,
                              hipStream_t stream) {
  const float* x   = (const float*)d_in[0];
  const float* Wh  = (const float*)d_in[1];
  const float* bh  = (const float*)d_in[2];
  const float* Wva = (const float*)d_in[3];
  const float* bva = (const float*)d_in[4];
  const float* Wvb = (const float*)d_in[5];
  const float* bvb = (const float*)d_in[6];
  const float* Wal = (const float*)d_in[7];
  const float* bal = (const float*)d_in[8];
  const float* Wbe = (const float*)d_in[9];
  const float* bbe = (const float*)d_in[10];

  char* ws = (char*)d_ws;
  _Float16* whp = (_Float16*)ws;                       // 786432 f16 = 1572864 B
  _Float16* wsp = (_Float16*)(ws + 1572864);           // 286720 f16 =  573440 B
  float*    bsp = (float*)(ws + 1572864 + 573440);     // 224 f32

  pack_wh  <<<3072, 256, 0, stream>>>(Wh, whp);
  pack_ws  <<<1120, 256, 0, stream>>>(Wva, Wvb, Wal, Wbe, wsp);
  pack_bias<<<1,    256, 0, stream>>>(bva, bvb, bal, bbe, bsp);

  nru_main<<<BATCH / BT, 256, LDS_TOTAL, stream>>>(x, whp, bh, wsp, bsp,
                                                   (float*)d_out);
}
